// SudokuLoss_73229192397510
// MI455X (gfx1250) — matrix operand order repaired and checked by reading, not compiled, not run on hardware
//
#include <hip/hip_runtime.h>
#include <hip/hip_bf16.h>

typedef __attribute__((ext_vector_type(16))) _Float16 v16h;
typedef __attribute__((ext_vector_type(8)))  float    v8f;

#define WAVES_PER_BLOCK 8
#define BLOCK_THREADS   256
#define NUM_BLOCKS      4096

// One wave32 per puzzle. Constraint sums D[27x11] = S[27x81] x P[81x11] via
// v_wmma_f32_16x16x32_f16 (S exact 0/1 in f16; column 10 of P carries the mask
// so the same WMMA yields per-group mask sums).
__global__ __launch_bounds__(BLOCK_THREADS) void sudoku_main(
    const float* __restrict__ logits,
    const int*   __restrict__ targets,
    const int*   __restrict__ puzzles,
    float*       __restrict__ ws,
    int B)
{
    // Per-wave FP16 tile, class-major: PT[class 0..15][cell 0..95]
    __shared__ __align__(64) _Float16 PT[WAVES_PER_BLOCK][16][96];

    const int lane  = threadIdx.x & 31;
    const int wid   = threadIdx.x >> 5;
    const int waveG = blockIdx.x * WAVES_PER_BLOCK + wid;
    const int waveT = gridDim.x * WAVES_PER_BLOCK;
    const int h     = lane >> 4;   // which 16-lane half
    const int nl    = lane & 15;   // N (class) for B/D, local M for A

    // Zero the wave's LDS tile once: pad cells 81..95 and class rows 11..15
    // stay zero for the whole kernel (only cells 0..80, rows 0..10 rewritten).
    {
        unsigned* p32 = (unsigned*)&PT[wid][0][0];
        for (int idx = lane; idx < (16 * 96) / 2; idx += 32) p32[idx] = 0u;
    }

    // Constant A fragments: S[group][cell], groups 0..8 rows, 9..17 cols,
    // 18..26 boxes. A uses contiguous K order here (same as B): lane half h, element e -> K =
    // e + 16*h.
    v16h afrag[2][3];
    #pragma unroll
    for (int a = 0; a < 2; ++a) {
        const int g = nl + 16 * a;
        #pragma unroll
        for (int t = 0; t < 3; ++t) {
            v16h f;
            #pragma unroll
            for (int e = 0; e < 16; ++e) {
                const int k = t * 32 + e + 16 * h;
                float v = 0.0f;
                if (k < 81 && g < 27) {
                    const int i = k / 9, j = k % 9;
                    int grp;
                    if (g < 9)       grp = i;
                    else if (g < 18) grp = 9 + j;
                    else             grp = 18 + (i / 3) * 3 + (j / 3);
                    v = (grp == g) ? 1.0f : 0.0f;
                }
                f[e] = (_Float16)v;
            }
            afrag[a][t] = f;
        }
    }
    __syncthreads();

    float ce_acc = 0.0f, mask_acc = 0.0f, sq_acc = 0.0f;
    const int iters = (B + waveT - 1) / waveT;   // uniform across all waves

    for (int it = 0; it < iters; ++it) {
        const int  b      = waveG + it * waveT;
        const bool active = (b < B);

        // -------- Phase A: softmax + CE, write p*mask (+mask) to LDS --------
        if (active) {
            for (int c = lane; c < 81; c += 32) {
                const float* lg = logits + ((size_t)b * 81 + c) * 10;
                float x[10];
                #pragma unroll
                for (int t = 0; t < 10; ++t) x[t] = __builtin_nontemporal_load(lg + t);
                float m = x[0];
                #pragma unroll
                for (int t = 1; t < 10; ++t) m = fmaxf(m, x[t]);
                float ex[10], s = 0.0f;
                #pragma unroll
                for (int t = 0; t < 10; ++t) { ex[t] = __expf(x[t] - m); s += ex[t]; }
                const float inv  = 1.0f / s;
                const int   tgt  = targets[(size_t)b * 81 + c];
                const float mask = (puzzles[(size_t)b * 81 + c] == 0) ? 1.0f : 0.0f;
                float xt = x[0];
                #pragma unroll
                for (int t = 1; t < 10; ++t) xt = (t == tgt) ? x[t] : xt;
                ce_acc   += mask * (m + __logf(s) - xt);  // -logp[target]
                mask_acc += mask;
                #pragma unroll
                for (int t = 0; t < 10; ++t)
                    PT[wid][t][c] = (_Float16)(ex[t] * inv * mask);
                PT[wid][10][c] = (_Float16)mask;
            }
        }
        __syncthreads();

        // -------- Phase B: D = S x P via 6 WMMAs (K = 96 in 3 steps) --------
        v8f acc0 = {};  // groups 0..15
        v8f acc1 = {};  // groups 16..26 (+pad)
        #pragma unroll
        for (int t = 0; t < 3; ++t) {
            // B in the same contiguous K order as A: lane half h holds K = t*32 + 16*h .. +15,
            // contiguous in class-major LDS -> one 32B aligned load.
            const v16h bfrag = *(const v16h*)&PT[wid][nl][t * 32 + 16 * h];
            acc0 = __builtin_amdgcn_wmma_f32_16x16x32_f16(
                       false, afrag[0][t], false, bfrag, (short)0, acc0, false, false);
            acc1 = __builtin_amdgcn_wmma_f32_16x16x32_f16(
                       false, afrag[1][t], false, bfrag, (short)0, acc1, false, false);
        }
        __syncthreads();

        // -------- Phase C: squared deviation from msum/9 (classes 1..9) -----
        // D layout: lane holds column n = lane&15; VGPR v holds row M = v + 8*h.
        // Column 10 = per-group mask sum -> broadcast within each 16-lane half.
        const bool nvalid = (nl >= 1 && nl <= 9);
        #pragma unroll
        for (int v = 0; v < 8; ++v) {
            const float ms0 = __shfl(acc0[v], (lane & 16) | 10, 32);
            const float ms1 = __shfl(acc1[v], (lane & 16) | 10, 32);
            if (active && nvalid) {
                const float d0 = acc0[v] - ms0 * (1.0f / 9.0f);
                sq_acc += d0 * d0;
                const int g1 = v + 8 * h + 16;
                if (g1 < 27) {
                    const float d1 = acc1[v] - ms1 * (1.0f / 9.0f);
                    sq_acc += d1 * d1;
                }
            }
        }
    }

    // Wave reduction, then 3 atomics per wave.
    #pragma unroll
    for (int off = 16; off > 0; off >>= 1) {
        ce_acc   += __shfl_xor(ce_acc,   off, 32);
        mask_acc += __shfl_xor(mask_acc, off, 32);
        sq_acc   += __shfl_xor(sq_acc,   off, 32);
    }
    if (lane == 0) {
        atomicAdd(&ws[0], ce_acc);
        atomicAdd(&ws[1], mask_acc);
        atomicAdd(&ws[2], sq_acc);
    }
}

__global__ void sudoku_init(float* ws) {
    if (threadIdx.x < 3) ws[threadIdx.x] = 0.0f;
}

__global__ void sudoku_finalize(const float* __restrict__ ws,
                                float* __restrict__ out, int B) {
    const float ce   = ws[0] / (ws[1] + 1e-8f);
    // row+col+box sq sums: each group_mse divides by B*(C-1)=B*9, then /27.
    const float cons = ws[2] / ((float)B * 9.0f * 27.0f);
    out[0] = ce + 0.1f * cons;   // total
    out[1] = ce;                 // ce_loss
    out[2] = cons;               // constraint_loss
}

extern "C" void kernel_launch(void* const* d_in, const int* in_sizes, int n_in,
                              void* d_out, int out_size, void* d_ws, size_t ws_size,
                              hipStream_t stream) {
    (void)n_in; (void)out_size; (void)ws_size;
    const float* logits  = (const float*)d_in[0];
    const int*   targets = (const int*)d_in[1];
    const int*   puzzles = (const int*)d_in[2];
    float*       out     = (float*)d_out;
    float*       ws      = (float*)d_ws;
    const int B = in_sizes[1] / 81;

    sudoku_init<<<1, 32, 0, stream>>>(ws);
    sudoku_main<<<NUM_BLOCKS, BLOCK_THREADS, 0, stream>>>(logits, targets, puzzles, ws, B);
    sudoku_finalize<<<1, 1, 0, stream>>>(ws, out, B);
}
